// DelayLIFSNN_87557203296275
// MI455X (gfx1250) — compile-verified, matrix operand order checked by reading, not run on hardware
//
#include <hip/hip_runtime.h>

typedef __attribute__((ext_vector_type(16))) _Float16 v16h;
typedef __attribute__((ext_vector_type(8)))  _Float16 v8h;
typedef __attribute__((ext_vector_type(8)))  float    v8f;

#define T0   300
#define BB   256
#define JJ   140
#define JP   160      // J padded to multiple of 32
#define HH   512
#define OO   20
#define OP   64       // O padded to one N tile
#define KD   25
#define LPAD 24
#define RPAD 12
#define BETA 0.95f
#define AP   40       // LDS row stride in halves (80B: 16B-aligned, conflict-free)

// ---------------------------------------------------------------------------
// CDNA5 async global->LDS copy (ASYNCcnt-tracked) and LDS fragment load.
// ---------------------------------------------------------------------------
__device__ __forceinline__ void async_copy_b128(const _Float16* gsrc, _Float16* ldst) {
    unsigned loff = (unsigned)(unsigned long long)ldst;  // low 32 bits = LDS offset
    asm volatile("global_load_async_to_lds_b128 %0, %1, off"
                 :: "v"(loff), "v"(gsrc) : "memory");
}

__device__ __forceinline__ v16h load_frag(const _Float16* p) {
    v8h lo = *(const v8h*)p;
    v8h hi = *(const v8h*)(p + 8);
    return __builtin_shufflevector(lo, hi, 0,1,2,3,4,5,6,7,8,9,10,11,12,13,14,15);
}

// ---------------------------------------------------------------------------
// Build time-padded f16 input: xpad (T0+36, B, JP); x is (B, T0, J) fp32.
// ---------------------------------------------------------------------------
__global__ void build_xpad_kernel(const float* __restrict__ x, _Float16* __restrict__ xpad) {
    long idx = (long)blockIdx.x * blockDim.x + threadIdx.x;
    const long total = (long)(T0 + LPAD + RPAD) * BB * JP;
    if (idx >= total) return;
    int i = (int)(idx % JP);
    long r = idx / JP;
    int b = (int)(r % BB);
    int t = (int)(r / BB);
    float v = 0.f;
    int ts = t - LPAD;
    if (i < JJ && ts >= 0 && ts < T0) v = x[((long)b * T0 + ts) * JJ + i];
    xpad[idx] = (_Float16)v;
}

// ---------------------------------------------------------------------------
// Build gaussian-delay kernel pack: kpack[n][k*CinPad + i] (f16), zero-padded.
// ---------------------------------------------------------------------------
__global__ void build_kpack_kernel(const float* __restrict__ w, const float* __restrict__ p,
                                   _Float16* __restrict__ kpack,
                                   int NoutPad, int Nreal, int Cin, int CinPad) {
    int idx = blockIdx.x * blockDim.x + threadIdx.x;
    if (idx >= NoutPad * CinPad) return;
    int n = idx / CinPad, i = idx % CinPad;
    long base = (long)n * CinPad * KD + i;
    if (n >= Nreal || i >= Cin) {
        for (int k = 0; k < KD; ++k) kpack[base + (long)k * CinPad] = (_Float16)0.f;
        return;
    }
    float wp = w[(long)n * Cin + i];
    float pp = p[(long)n * Cin + i];
    float g[KD];
    float s = 0.f;
    for (int k = 0; k < KD; ++k) {
        float d = ((float)k - (float)(KD / 2) - pp) * 2.0f;  // /SIG, SIG=0.5
        float e = __expf(-0.5f * d * d);
        g[k] = e; s += e;
    }
    float inv = 1.f / (s + 1e-7f);
    for (int k = 0; k < KD; ++k)
        kpack[base + (long)k * CinPad] = (_Float16)(wp * g[k] * inv);
}

// ---------------------------------------------------------------------------
// Delay-conv as shifted WMMA GEMM with async-LDS double buffering.
//   out[t, b, n] = sum_k sum_i xpad[t+k, b, i] * kpack[n, k*Cin+i]
// Block = 8 waves -> 256(M) x 128(N) tile at time t = blockIdx.y.
// ---------------------------------------------------------------------------
__global__ void __launch_bounds__(256)
wmma_delay_gemm_lds(const _Float16* __restrict__ xpad,   // (Tin, BB, Cin) f16
                    const _Float16* __restrict__ kpack,  // (N, Cin*KD)    f16
                    float* __restrict__ out,             // (Tout, BB, N)  f32
                    int Cin, int N) {
    __shared__ __align__(128) _Float16 Ash[2][BB][AP];    // 40 KB
    __shared__ __align__(128) _Float16 Bsh[2][128][AP];   // 20 KB

    const int t    = blockIdx.y;
    const int n0b  = blockIdx.x << 7;
    const int tid  = threadIdx.x;
    const int wave = tid >> 5;
    const int lane = tid & 31;
    const int lh = lane & 15, hi = lane >> 4;
    const int tm = wave & 3, tn = wave >> 2;
    const int m0w = tm << 6, n0w = tn << 6;
    const long ldk = (long)Cin * KD;
    const int seg = tid & 3;   // 16B segment within a 64B row payload
    const int rb  = tid >> 2;  // 0..63

    const int nI = Cin >> 5;
    const int C  = KD * nI;

    v8f acc[4][4] = {};

    auto stage = [&](int c, int buf) {
        int k = c / nI;
        int i0 = (c - k * nI) << 5;
        const _Float16* ga = xpad + (long)(t + k) * BB * Cin + i0 + seg * 8;
        const _Float16* gb = kpack + (long)n0b * ldk + (long)k * Cin + i0 + seg * 8;
        _Float16* la = &Ash[buf][0][0] + seg * 8;
        _Float16* lb = &Bsh[buf][0][0] + seg * 8;
#pragma unroll
        for (int j = 0; j < 4; ++j) {
            int r = j * 64 + rb;
            async_copy_b128(ga + (long)r * Cin, la + r * AP);
        }
#pragma unroll
        for (int j = 0; j < 2; ++j) {
            int cc = j * 64 + rb;
            async_copy_b128(gb + (long)cc * ldk, lb + cc * AP);
        }
    };

    stage(0, 0);
    for (int c = 0; c < C; ++c) {
        const int buf = c & 1;
        if (c + 1 < C) {
            stage(c + 1, buf ^ 1);
            // drain only the PREVIOUS chunk's 6 per-thread async ops
            asm volatile("s_wait_asynccnt 6" ::: "memory");
        } else {
            asm volatile("s_wait_asynccnt 0" ::: "memory");
        }
        __syncthreads();   // current buffer complete across all waves

        v16h afrag[4], bfrag[4];
#pragma unroll
        for (int mi = 0; mi < 4; ++mi)
            afrag[mi] = load_frag(&Ash[buf][m0w + mi * 16 + lh][hi * 16]);
#pragma unroll
        for (int ni = 0; ni < 4; ++ni)
            bfrag[ni] = load_frag(&Bsh[buf][n0w + ni * 16 + lh][hi * 16]);
#pragma unroll
        for (int mi = 0; mi < 4; ++mi)
#pragma unroll
            for (int ni = 0; ni < 4; ++ni)
                acc[mi][ni] = __builtin_amdgcn_wmma_f32_16x16x32_f16(
                    false, afrag[mi], false, bfrag[ni],
                    (short)0, acc[mi][ni], false, false);

        __syncthreads();   // buffer free before it is re-staged next iteration
    }

    // C/D layout: VGPR j of lane: row = 16*mi + 8*hi + j, col = 16*ni + lh
    float* obase = out + (long)t * BB * N;
#pragma unroll
    for (int mi = 0; mi < 4; ++mi)
#pragma unroll
        for (int ni = 0; ni < 4; ++ni)
#pragma unroll
            for (int j = 0; j < 8; ++j) {
                int row = m0w + mi * 16 + hi * 8 + j;
                int col = n0b + n0w + ni * 16 + lh;
                obase[(long)row * N + col] = acc[mi][ni][j];
            }
}

// ---------------------------------------------------------------------------
// Direct-load WMMA GEMM (used for the small readout conv, N=64).
// ---------------------------------------------------------------------------
__global__ void __launch_bounds__(256)
wmma_delay_gemm(const _Float16* __restrict__ xpad,
                const _Float16* __restrict__ kpack,
                float* __restrict__ out,
                int Cin, int N) {
    const int t    = blockIdx.y;
    const int wave = threadIdx.x >> 5;
    const int lane = threadIdx.x & 31;
    const int mt = BB >> 6;
    const int nt = N  >> 6;
    const int tile = blockIdx.x * 8 + wave;
    if (tile >= mt * nt) return;
    const int tm = tile % mt, tn = tile / mt;
    const int m0 = tm << 6, n0 = tn << 6;
    const int lh = lane & 15, hi = lane >> 4;
    const long ldk = (long)Cin * KD;

    v8f acc[4][4] = {};

    for (int k = 0; k < KD; ++k) {
        const _Float16* abase = xpad + ((long)(t + k) * BB + m0) * Cin + hi * 16;
        const _Float16* bbase = kpack + (long)(n0 + lh) * ldk + (long)k * Cin + hi * 16;
        for (int i0 = 0; i0 < Cin; i0 += 32) {
            v16h afrag[4];
#pragma unroll
            for (int mi = 0; mi < 4; ++mi)
                afrag[mi] = *(const v16h*)(abase + i0 + (long)(mi * 16 + lh) * Cin);
            v16h bfrag[4];
#pragma unroll
            for (int ni = 0; ni < 4; ++ni)
                bfrag[ni] = *(const v16h*)(bbase + i0 + (long)(ni * 16) * ldk);
#pragma unroll
            for (int mi = 0; mi < 4; ++mi)
#pragma unroll
                for (int ni = 0; ni < 4; ++ni)
                    acc[mi][ni] = __builtin_amdgcn_wmma_f32_16x16x32_f16(
                        false, afrag[mi], false, bfrag[ni],
                        (short)0, acc[mi][ni], false, false);
        }
    }

    float* obase = out + (long)t * BB * N;
#pragma unroll
    for (int mi = 0; mi < 4; ++mi)
#pragma unroll
        for (int ni = 0; ni < 4; ++ni)
#pragma unroll
            for (int j = 0; j < 8; ++j) {
                int row = m0 + mi * 16 + hi * 8 + j;
                int col = n0 + ni * 16 + lh;
                obase[(long)row * N + col] = acc[mi][ni][j];
            }
}

// ---------------------------------------------------------------------------
// BatchNorm stats: per-channel sum / sumsq over rows (T*B), coalesced.
// ---------------------------------------------------------------------------
__global__ void bn_stats_kernel(const float* __restrict__ conv, float* __restrict__ sums,
                                long rows, int rowsPerBlock) {
    int h = threadIdx.x;  // blockDim.x == HH
    long r0 = (long)blockIdx.x * rowsPerBlock;
    long r1 = r0 + rowsPerBlock; if (r1 > rows) r1 = rows;
    float s = 0.f, s2 = 0.f;
    for (long r = r0; r < r1; ++r) {
        float v = conv[r * HH + h];
        s += v; s2 += v * v;
    }
    atomicAdd(&sums[h], s);
    atomicAdd(&sums[HH + h], s2);
}

__global__ void bn_finalize_kernel(const float* __restrict__ sums,
                                   const float* __restrict__ gamma, const float* __restrict__ beta,
                                   float* __restrict__ scale, float* __restrict__ shift, float invN) {
    int h = blockIdx.x * blockDim.x + threadIdx.x;
    if (h >= HH) return;
    float mu  = sums[h] * invN;
    float var = sums[HH + h] * invN - mu * mu;
    float sc  = gamma[h] * rsqrtf(var + 1e-5f);
    scale[h] = sc;
    shift[h] = beta[h] - mu * sc;
}

// ---------------------------------------------------------------------------
// LIF scan: U = U*beta + (y*scale+shift); S = (U>=1); U = (1-S)*U.
// Writes f16 spikes into the (time-padded) buffer at row t+LPAD.
// ---------------------------------------------------------------------------
__global__ void lif_kernel(const float* __restrict__ conv,
                           const float* __restrict__ scale, const float* __restrict__ shift,
                           _Float16* __restrict__ spad, int T) {
    int idx = blockIdx.x * blockDim.x + threadIdx.x;
    const int BH = BB * HH;
    if (idx >= BH) return;
    int h = idx & (HH - 1);
    float sc = scale[h], sh = shift[h];
    float U = 0.f;
    for (int t = 0; t < T; ++t) {
        float v = conv[(long)t * BH + idx] * sc + sh;
        U = U * BETA + v;
        float S = (U >= 1.0f) ? 1.f : 0.f;
        spad[(long)(t + LPAD) * BH + idx] = (_Float16)S;
        U = (1.f - S) * U;
    }
}

// ---------------------------------------------------------------------------
// Readout: LI scan over t, softmax over O, sum over t. One thread per batch.
// ---------------------------------------------------------------------------
__global__ void readout_kernel(const float* __restrict__ convr, float* __restrict__ out) {
    int b = blockIdx.x * blockDim.x + threadIdx.x;
    if (b >= BB) return;
    float U[OO], acc[OO];
    for (int o = 0; o < OO; ++o) { U[o] = 0.f; acc[o] = 0.f; }
    const int T3 = T0 + RPAD + RPAD + RPAD;  // 336
    for (int t = 0; t < T3; ++t) {
        const float* row = convr + ((long)t * BB + b) * OP;
        float mx = -1e30f;
        for (int o = 0; o < OO; ++o) {
            U[o] = U[o] * BETA + row[o];
            mx = fmaxf(mx, U[o]);
        }
        float se = 0.f;
        float e[OO];
        for (int o = 0; o < OO; ++o) { e[o] = __expf(U[o] - mx); se += e[o]; }
        float inv = 1.f / se;
        for (int o = 0; o < OO; ++o) acc[o] += e[o] * inv;
    }
    for (int o = 0; o < OO; ++o) out[(long)b * OO + o] = acc[o];
}

// ---------------------------------------------------------------------------
extern "C" void kernel_launch(void* const* d_in, const int* in_sizes, int n_in,
                              void* d_out, int out_size, void* d_ws, size_t ws_size,
                              hipStream_t stream) {
    const float* x  = (const float*)d_in[0];
    const float* w0 = (const float*)d_in[1];
    const float* p0 = (const float*)d_in[2];
    const float* g0 = (const float*)d_in[3];
    const float* b0 = (const float*)d_in[4];
    const float* w1 = (const float*)d_in[5];
    const float* p1 = (const float*)d_in[6];
    const float* g1 = (const float*)d_in[7];
    const float* b1 = (const float*)d_in[8];
    const float* wr = (const float*)d_in[9];
    const float* pr = (const float*)d_in[10];

    const int T1 = T0 + RPAD;        // 312: layer0 conv out
    const int T2 = T1 + RPAD;        // 324: layer1 conv out
    const int T3 = T2 + RPAD;        // 336: readout conv out

    char* ws = (char*)d_ws;
    size_t off = 0;
    auto alloc = [&](size_t bytes) -> char* {
        char* p = ws + off;
        off = (off + bytes + 255) & ~(size_t)255;
        return p;
    };
    _Float16* xpad = (_Float16*)alloc((size_t)(T0 + 36) * BB * JP * 2);   // 27.5 MB
    _Float16* kp0  = (_Float16*)alloc((size_t)HH * JP * KD * 2);          //  4.1 MB
    _Float16* kp1  = (_Float16*)alloc((size_t)HH * HH * KD * 2);          // 13.1 MB
    _Float16* kpr  = (_Float16*)alloc((size_t)OP * HH * KD * 2);          //  1.6 MB
    float*    conv = (float*)   alloc((size_t)T2 * BB * HH * 4);          // 170 MB (reused 3x)
    _Float16* s0   = (_Float16*)alloc((size_t)(T1 + 36) * BB * HH * 2);   // 91 MB
    _Float16* s1   = (_Float16*)alloc((size_t)(T2 + 36) * BB * HH * 2);   // 94 MB
    float*    sums = (float*)   alloc(2 * HH * 4);
    float*    scl  = (float*)   alloc(HH * 4);
    float*    shf  = (float*)   alloc(HH * 4);

    const long BH = (long)BB * HH;
    // Zero only the time-pad regions of the spike buffers.
    hipMemsetAsync(s0, 0, (size_t)LPAD * BH * 2, stream);
    hipMemsetAsync(s0 + (long)(LPAD + T1) * BH, 0, (size_t)RPAD * BH * 2, stream);
    hipMemsetAsync(s1, 0, (size_t)LPAD * BH * 2, stream);
    hipMemsetAsync(s1 + (long)(LPAD + T2) * BH, 0, (size_t)RPAD * BH * 2, stream);

    // Stage 0: pack inputs & kernels to f16
    {
        long total = (long)(T0 + 36) * BB * JP;
        build_xpad_kernel<<<dim3((unsigned)((total + 255) / 256)), 256, 0, stream>>>(x, xpad);
        build_kpack_kernel<<<dim3((HH * JP + 255) / 256), 256, 0, stream>>>(w0, p0, kp0, HH, HH, JJ, JP);
        build_kpack_kernel<<<dim3((HH * HH + 255) / 256), 256, 0, stream>>>(w1, p1, kp1, HH, HH, HH, HH);
        build_kpack_kernel<<<dim3((OP * HH + 255) / 256), 256, 0, stream>>>(wr, pr, kpr, OP, OO, HH, HH);
    }

    // Layer 0: delay conv (WMMA + async LDS) -> BN -> LIF
    wmma_delay_gemm_lds<<<dim3(HH / 128, T1), 256, 0, stream>>>(xpad, kp0, conv, JP, HH);
    hipMemsetAsync(sums, 0, 2 * HH * 4, stream);
    bn_stats_kernel<<<dim3(T1), HH, 0, stream>>>(conv, sums, (long)T1 * BB, BB);
    bn_finalize_kernel<<<dim3(2), 256, 0, stream>>>(sums, g0, b0, scl, shf, 1.0f / ((float)T1 * BB));
    lif_kernel<<<dim3((unsigned)(BH / 256)), 256, 0, stream>>>(conv, scl, shf, s0, T1);

    // Layer 1: delay conv (WMMA + async LDS) -> BN -> LIF
    wmma_delay_gemm_lds<<<dim3(HH / 128, T2), 256, 0, stream>>>(s0, kp1, conv, HH, HH);
    hipMemsetAsync(sums, 0, 2 * HH * 4, stream);
    bn_stats_kernel<<<dim3(T2), HH, 0, stream>>>(conv, sums, (long)T2 * BB, BB);
    bn_finalize_kernel<<<dim3(2), 256, 0, stream>>>(sums, g1, b1, scl, shf, 1.0f / ((float)T2 * BB));
    lif_kernel<<<dim3((unsigned)(BH / 256)), 256, 0, stream>>>(conv, scl, shf, s1, T2);

    // Readout: delay conv (WMMA direct, O padded to 64) -> LI -> softmax -> sum
    wmma_delay_gemm<<<dim3(1, T3), 256, 0, stream>>>(s1, kpr, conv, HH, OP);
    readout_kernel<<<dim3(1), 256, 0, stream>>>(conv, (float*)d_out);

    (void)in_sizes; (void)n_in; (void)out_size; (void)ws_size;
}
